// RayObsGraph_51479478009938
// MI455X (gfx1250) — compile-verified, matrix operand order checked by reading, not run on hardware
//
#include <hip/hip_runtime.h>

// RayObsGraph fused forward for MI455X (gfx1250), wave32 + WMMA bf16.
//
// prep_weights (once per launch): W0_rel/W0_root f32 -> bf16, pre-swizzled
//   into WMMA B-fragment layout in d_ws (shared by all 256 batch blocks).
// rayobs_fused (256 blocks x 1024 threads = 32 waves):
//   x     = nodes[b] with row n := flat[b]                    [128,1024]
//   one K-pass computes BOTH x@W0_rel and x@W0_root (8 acc tiles/wave);
//   W tiles arrive via global_load_async_to_lds_b128 (ASYNCcnt),
//   X tiles via packed bf16 converts into swizzled LDS.
//   h0    = relu(A^T @ Y_rel + x @ W0_root + b0)              [128,256]
//   aggr1 = A^T @ h0 (WMMA again); v = aggr1[n]; u = h0[n]
//   h1row = relu(v @ W1_rel + u @ W1_root + b1)               [1024]
//   logits = h1row @ Wl + bl ; value = h1row @ Wv + bv

#define GSZ   128
#define OBS   1024
#define HDIM  256
#define OUTD  1024
#define NOUT  18
#define BATCH 256
#define NK    (OBS / 32)

typedef __attribute__((ext_vector_type(16))) __bf16 v16bf;
typedef __attribute__((ext_vector_type(2)))  __bf16 v2bf;
typedef __attribute__((ext_vector_type(8)))  float  v8f;

// Packed f32x2 -> bf16x2 via compiler-native conversion (backend selects
// v_cvt_(pk_)bf16_f32 on gfx1250, or inlines RNE otherwise).
__device__ __forceinline__ unsigned int pkbf(float lo, float hi) {
  v2bf p;
  p[0] = (__bf16)lo;
  p[1] = (__bf16)hi;
  union { v2bf v; unsigned int u; } c; c.v = p; return c.u;
}
__device__ __forceinline__ float bf2f(unsigned short u) {
  union { unsigned int u; float f; } c; c.u = ((unsigned int)u) << 16;
  return c.f;
}
__device__ __forceinline__ v8f zero8() {
  v8f z = {0.f, 0.f, 0.f, 0.f, 0.f, 0.f, 0.f, 0.f};
  return z;
}

// Offset of element (mn, k) inside a 512-element swizzled fragment.
// ISA 7.12.2 (16-bit A/B): lane = (mn&15) + 16*((k>>3)&1),
//                          idx  = 8*((k>>4)&1) + (k&7); 16 contig per lane.
__device__ __forceinline__ int swz(int mn, int k) {
  int lane = (mn & 15) | (((k >> 3) & 1) << 4);
  int idx  = (((k >> 4) & 1) << 3) | (k & 7);
  return (lane << 4) | idx;
}
// Element (col c, K-index j) inside a [4 kc][16 nt][512] B-fragment buffer.
__device__ __forceinline__ int yoff(int c, int j) {
  return (((j >> 5) << 4) + (c >> 4)) * 512 + swz(c, j & 31);
}

// One fragment (16 bf16) per lane, contiguous: one 32B LDS read.
__device__ __forceinline__ v16bf read_frag(const unsigned short* fragbase) {
  int lane = threadIdx.x & 31;
  return *reinterpret_cast<const v16bf*>(fragbase + (lane << 4));
}

// Pack one C/D tile column (8 rows, contiguous in the B-swizzle) and store
// it as a single 16B LDS write.
__device__ __forceinline__ void store_tile_frag(unsigned short* dstbase,
                                                int nt, int crow, int col,
                                                const v8f acc) {
  uint4 pk;
  pk.x = pkbf(acc[0], acc[1]);
  pk.y = pkbf(acc[2], acc[3]);
  pk.z = pkbf(acc[4], acc[5]);
  pk.w = pkbf(acc[6], acc[7]);
  unsigned short* dst = dstbase + (((crow >> 5) << 4) + nt) * 512 + swz(col, crow & 31);
  *(uint4*)dst = pk;
}

// Async DMA 16B global -> LDS (GLOBAL_LOAD_ASYNC_TO_LDS_B128, ASYNCcnt).
__device__ __forceinline__ void async_g2l_16B(unsigned short* l, const unsigned short* g) {
  unsigned lds_off = (unsigned)(size_t)l;
  unsigned long long ga = (unsigned long long)(size_t)g;
  asm volatile("global_load_async_to_lds_b128 %0, %1, off"
               :: "v"(lds_off), "v"(ga) : "memory");
}
__device__ __forceinline__ void wait_async0() {
  asm volatile("s_wait_asynccnt 0" ::: "memory");
}

// ---------------- pre-pass: weights -> swizzled bf16 fragments ----------
// ws layout (ushort): [m(2)][kc(32)][nt(16)][512];  m=0: W0_rel, m=1: W0_root
__global__ __launch_bounds__(1024)
void prep_weights(const float* __restrict__ W0_rel,
                  const float* __restrict__ W0_root,
                  unsigned short* __restrict__ wsz) {
  int idx = blockIdx.x * 1024 + threadIdx.x;   // 0 .. 262143 (k-pairs x cols)
  int m   = idx >> 17;
  int rem = idx & 131071;
  int kp  = rem >> 8, col = rem & 255, k = kp << 1;
  const float* W = m ? W0_root : W0_rel;
  float lo = W[k * HDIM + col];
  float hi = W[(k + 1) * HDIM + col];
  int off = ((((m << 5) + (k >> 5)) << 4) + (col >> 4)) * 512 + swz(col, k & 31);
  *(unsigned int*)(wsz + off) = pkbf(lo, hi);  // even idx -> 4B aligned
}

// Stage K-chunk kc: X [128x32] (row n := flat, pair-convert) and both W
// matrices' pre-swizzled chunks (pure async 16B copies, no VALU).
__device__ __forceinline__ void stage_chunk(int kc,
                                            const unsigned short* __restrict__ wsz,
                                            const float* __restrict__ nodes_b,
                                            const float* __restrict__ flat_b,
                                            int n,
                                            unsigned short* __restrict__ sx,
                                            unsigned short* __restrict__ sw) {
#pragma unroll
  for (int i = 0; i < 2; ++i) {                 // 2048 k-pairs / 1024 threads
    int e = threadIdx.x + (i << 10);
    int r = e >> 4, k = (e & 15) << 1;
    const float* src = (r == n) ? (flat_b + (kc << 5) + k)
                                : (nodes_b + r * OBS + (kc << 5) + k);
    float2 xv = *(const float2*)src;
    *(unsigned int*)(sx + ((r >> 4) << 9) + swz(r, k)) = pkbf(xv.x, xv.y);
  }
#pragma unroll
  for (int m = 0; m < 2; ++m) {                 // 1024 16B units per matrix
    int u = threadIdx.x;
    async_g2l_16B(sw + (m << 13) + (u << 3),
                  wsz + ((((m << 5) + kc)) << 13) + (u << 3));
  }
}

__global__ __launch_bounds__(1024)
void rayobs_fused(const float* __restrict__ flat,
                  const float* __restrict__ nodes,
                  const int* __restrict__ num_nodes,
                  const int* __restrict__ adj,
                  const unsigned short* __restrict__ wsz,  // swizzled W0 bf16
                  const float* __restrict__ b0,
                  const float* __restrict__ W1_rel, const float* __restrict__ b1,
                  const float* __restrict__ W1_root,
                  const float* __restrict__ Wl, const float* __restrict__ bl,
                  const float* __restrict__ Wv, const float* __restrict__ bv,
                  float* __restrict__ out) {
  // A^T in A-fragment layout: [8 mt][4 kc][512]                     (32 KB)
  __shared__ __attribute__((aligned(16))) unsigned short s_at[8 * 4 * 512];
  // Y_rel, then h0, both in B-fragment layout [4 kc][16 nt][512]    (64 KB)
  __shared__ __attribute__((aligned(16))) unsigned short s_y[4 * 16 * 512];
  // Double-buffered staging: X A-frags (16 KB), W B-frags x2 mats   (64 KB)
  __shared__ __attribute__((aligned(16))) unsigned short s_x[2 * 8 * 512];
  __shared__ __attribute__((aligned(16))) unsigned short s_w[2 * 2 * 16 * 512];
  __shared__ float s_v[HDIM];
  __shared__ float s_u[HDIM];
  __shared__ float s_h1[OUTD];

  const int b = blockIdx.x;
  const int n = num_nodes[b];
  const float* flat_b  = flat  + (size_t)b * OBS;
  const float* nodes_b = nodes + (size_t)b * GSZ * OBS;
  const int*   adj_b   = adj   + (size_t)b * GSZ * GSZ;

  const int wid  = threadIdx.x >> 5;
  const int lane = threadIdx.x & 31;
  const int mt   = wid & 7;
  const int ntb  = (wid >> 3) * 4;

  // ---- Build A^T (binarized, swizzled, pair-packed) + injected edges ----
#pragma unroll
  for (int i = 0; i < 8; ++i) {                 // 8192 j-pairs / 1024 threads
    int p = threadIdx.x + (i << 10);
    int jp = p >> 7, ii = p & 127, j = jp << 1;
    int nz0 = (adj_b[j * GSZ + ii] != 0);
    int nz1 = (adj_b[(j + 1) * GSZ + ii] != 0);
    unsigned int pk = (nz0 ? 0x3F80u : 0u) | (nz1 ? 0x3F800000u : 0u);
    *(unsigned int*)(s_at + (((ii >> 4) << 2) + (j >> 5)) * 512 + swz(ii, j & 31)) = pk;
  }
  __syncthreads();
  if (threadIdx.x == 0) {
    s_at[(((n >> 4) << 2) + (n >> 5)) * 512 + swz(n, n & 31)] = 0x3F80;   // At[n][n]
    if (n > 0) {
      int p = n - 1;
      s_at[(((p >> 4) << 2) + (n >> 5)) * 512 + swz(p, n & 31)] = 0x3F80; // At[n-1][n]
      s_at[(((n >> 4) << 2) + (p >> 5)) * 512 + swz(n, p & 31)] = 0x3F80; // At[n][n-1]
    }
  }
  __syncthreads();

  const int rofs  = (lane >> 4) << 3;      // C/D layout: M offset 0 or 8
  const int crow  = mt * 16 + rofs;
  const int ccol0 = lane & 15;

  // ---- One K-pass: accR = X@W0_rel, accO = X@W0_root ----
  v8f accR[4], accO[4];
#pragma unroll
  for (int t = 0; t < 4; ++t) { accR[t] = zero8(); accO[t] = zero8(); }

  stage_chunk(0, wsz, nodes_b, flat_b, n, s_x, s_w);
  wait_async0();
  __syncthreads();
  for (int kc = 0; kc < NK; ++kc) {
    int cur = kc & 1;
    const unsigned short* sxc = s_x + cur * (8 * 512);
    const unsigned short* swc = s_w + cur * (2 * 16 * 512);
    if (kc + 1 < NK) {
      __builtin_prefetch(wsz + ((size_t)(kc + 1) << 13), 0, 1);
      __builtin_prefetch(wsz + ((size_t)(32 + kc + 1) << 13), 0, 1);
      stage_chunk(kc + 1, wsz, nodes_b, flat_b, n,
                  s_x + (cur ^ 1) * (8 * 512), s_w + (cur ^ 1) * (2 * 16 * 512));
    }
    v16bf a = read_frag(sxc + (mt << 9));
#pragma unroll
    for (int t = 0; t < 4; ++t) {
      v16bf bb = read_frag(swc + ((ntb + t) << 9));
      accR[t] = __builtin_amdgcn_wmma_f32_16x16x32_bf16(
          false, a, false, bb, (short)0, accR[t], false, false);
    }
#pragma unroll
    for (int t = 0; t < 4; ++t) {
      v16bf bb = read_frag(swc + ((16 + ntb + t) << 9));
      accO[t] = __builtin_amdgcn_wmma_f32_16x16x32_bf16(
          false, a, false, bb, (short)0, accO[t], false, false);
    }
    wait_async0();
    __syncthreads();
  }

  // ---- Y_rel -> s_y (B-fragment layout), 1 b128 store per tile ----
#pragma unroll
  for (int t = 0; t < 4; ++t) {
    int nt = ntb + t, col = (nt << 4) + ccol0;
    store_tile_frag(s_y, nt, crow, col, accR[t]);
  }
  __syncthreads();

  // ---- accO += A^T @ Y_rel (K = 128, all fragments pre-swizzled) ----
#pragma unroll
  for (int kc = 0; kc < 4; ++kc) {
    v16bf a = read_frag(s_at + (((mt << 2) + kc) << 9));
#pragma unroll
    for (int t = 0; t < 4; ++t) {
      v16bf bb = read_frag(s_y + (((kc << 4) + ntb + t) << 9));
      accO[t] = __builtin_amdgcn_wmma_f32_16x16x32_bf16(
          false, a, false, bb, (short)0, accO[t], false, false);
    }
  }
  __syncthreads();  // all Y_rel reads done before h0 overwrites s_y

  // ---- h0 = relu(accO + b0) -> s_y (B-fragment layout again) ----
#pragma unroll
  for (int t = 0; t < 4; ++t) {
    int nt = ntb + t, col = (nt << 4) + ccol0;
    float bias = b0[col];
    v8f hv;
#pragma unroll
    for (int r = 0; r < 8; ++r) {
      float x = accO[t][r] + bias;
      hv[r] = x > 0.f ? x : 0.f;
    }
    store_tile_frag(s_y, nt, crow, col, hv);
  }
  __syncthreads();

  // ---- aggr1 = A^T @ h0 (WMMA); v = aggr1[n]; u = h0[n] ----
  v8f accA[4];
#pragma unroll
  for (int t = 0; t < 4; ++t) accA[t] = zero8();
#pragma unroll
  for (int kc = 0; kc < 4; ++kc) {
    v16bf a = read_frag(s_at + (((mt << 2) + kc) << 9));
#pragma unroll
    for (int t = 0; t < 4; ++t) {
      v16bf bb = read_frag(s_y + (((kc << 4) + ntb + t) << 9));
      accA[t] = __builtin_amdgcn_wmma_f32_16x16x32_bf16(
          false, a, false, bb, (short)0, accA[t], false, false);
    }
  }
  // Waves holding M-tile n>>4 export row n of aggr1 (f32, no round-trip).
  if (mt == (n >> 4) && (lane >> 4) == ((n >> 3) & 1)) {
    int r = n & 7;
#pragma unroll
    for (int t = 0; t < 4; ++t)
      s_v[((ntb + t) << 4) + (lane & 15)] = accA[t][r];
  }
  if (threadIdx.x < HDIM) {
    int c = threadIdx.x;
    s_u[c] = bf2f(s_y[yoff(c, n)]);
  }
  __syncthreads();

  // ---- h1row = relu(v @ W1_rel + u @ W1_root + b1) ----
  {
    int o = threadIdx.x;  // 1024 threads = 1024 outputs, coalesced W reads
    float a1 = b1[o];
#pragma unroll 4
    for (int c = 0; c < HDIM; ++c)
      a1 += s_v[c] * W1_rel[c * OUTD + o] + s_u[c] * W1_root[c * OUTD + o];
    s_h1[o] = a1 > 0.f ? a1 : 0.f;
  }
  __syncthreads();

  // ---- logits / value: one wave32 per output, shfl_xor reduce ----
  if (wid < NOUT + 1) {
    float a1 = 0.f;
    if (wid < NOUT) {
      for (int o = lane; o < OUTD; o += 32) a1 += s_h1[o] * Wl[o * NOUT + wid];
    } else {
      for (int o = lane; o < OUTD; o += 32) a1 += s_h1[o] * Wv[o];
    }
#pragma unroll
    for (int off = 16; off > 0; off >>= 1) a1 += __shfl_xor(a1, off, 32);
    if (lane == 0) {
      if (wid < NOUT) out[b * NOUT + wid] = a1 + bl[wid];
      else            out[BATCH * NOUT + b] = a1 + bv[0];
    }
  }
}

extern "C" void kernel_launch(void* const* d_in, const int* in_sizes, int n_in,
                              void* d_out, int out_size, void* d_ws, size_t ws_size,
                              hipStream_t stream) {
  (void)in_sizes; (void)n_in; (void)out_size; (void)ws_size;
  const float* flat     = (const float*)d_in[0];
  const float* nodes    = (const float*)d_in[1];
  const int*   num_node = (const int*)d_in[2];
  const int*   adj      = (const int*)d_in[3];
  // d_in[4] = seq_lens (unused; only fixes B)
  const float* W0_rel   = (const float*)d_in[5];
  const float* b0       = (const float*)d_in[6];
  const float* W0_root  = (const float*)d_in[7];
  const float* W1_rel   = (const float*)d_in[8];
  const float* b1       = (const float*)d_in[9];
  const float* W1_root  = (const float*)d_in[10];
  const float* Wl       = (const float*)d_in[11];
  const float* bl       = (const float*)d_in[12];
  const float* Wv       = (const float*)d_in[13];
  const float* bv       = (const float*)d_in[14];
  float* out = (float*)d_out;
  unsigned short* wsz = (unsigned short*)d_ws;  // needs 1 MB scratch

  prep_weights<<<256, 1024, 0, stream>>>(W0_rel, W0_root, wsz);
  rayobs_fused<<<BATCH, 1024, 0, stream>>>(flat, nodes, num_node, adj,
                                           wsz, b0,
                                           W1_rel, b1, W1_root,
                                           Wl, bl, Wv, bv, out);
}
